// TSNet_51496657879791
// MI455X (gfx1250) — compile-verified, matrix operand order checked by reading, not run on hardware
//
#include <hip/hip_runtime.h>

typedef __attribute__((ext_vector_type(16))) __bf16 v16bf;
typedef __attribute__((ext_vector_type(8)))  float  v8f;

union Frag {
    uint4  u[2];
    v16bf  v;
};

__device__ __forceinline__ unsigned short f2bf(float f) {
    unsigned int u = __float_as_uint(f);
    unsigned int r = u + 0x7FFFu + ((u >> 16) & 1u);   // round-to-nearest-even
    return (unsigned short)(r >> 16);
}

__device__ __forceinline__ unsigned int pack2bf(float lo, float hi) {
    return (unsigned int)f2bf(lo) | ((unsigned int)f2bf(hi) << 16);
}

#define D_DIM 3328
#define H_DIM 4096
#define C_DIM 200
#define S_DIM 8
#define B_DIM 2048
#define M_DIM (B_DIM * S_DIM)   // 16384

#define BK   64
#define BKP  72                 // padded LDS stride (144B: 16B-aligned rows, bank spread)

// ---------------------------------------------------------------------------
// Kernel 1: z = relu( (x@W1 + b1 - mean) * gamma*rsqrt(var+eps) + beta ), bf16
// WG tile 128(M) x 256(N), BK=64. 8 waves as 2(row) x 4(col); wave tile 64x64
// = 4x4 WMMA tiles -> 16 ds_load_b128 per 16 v_wmma per k-step.
// ---------------------------------------------------------------------------
__global__ __launch_bounds__(256) void gemm1_bn_relu(
    const float* __restrict__ x, const float* __restrict__ W1,
    const float* __restrict__ b1, const float* __restrict__ gam,
    const float* __restrict__ bet, const float* __restrict__ mu,
    const float* __restrict__ var, unsigned short* __restrict__ z)
{
    __shared__ unsigned short ldsA[128 * BKP];
    __shared__ unsigned short ldsB[256 * BKP];

    const int t    = threadIdx.x;
    const int wgN  = blockIdx.x;      // 0..15  (H tiles of 256) — x-fastest => W1 L2-hot
    const int wgM  = blockIdx.y;      // 0..127 (row tiles of 128)
    const int wave = t >> 5, lane = t & 31, hi = lane >> 4, lm = lane & 15;
    const int wr = wave >> 2, wc = wave & 3;      // 2 x 4 wave grid

    v8f acc[4][4];
    #pragma unroll
    for (int i = 0; i < 4; ++i)
        #pragma unroll
        for (int j = 0; j < 4; ++j)
            #pragma unroll
            for (int e = 0; e < 8; ++e) acc[i][j][e] = 0.0f;

    // staging coordinates
    const int rrA = t >> 4;            // 0..15 (A row group)
    const int kcA = (t & 15) * 4;      // A col (k) within tile
    const int n2  = (t & 127) * 2;     // B: two adjacent columns
    const int kbB = (t >> 7) * 2;      // B: k phase 0 or 2

    for (int k0 = 0; k0 < D_DIM; k0 += BK) {
        // ---- stage A tile (128 x 64) fp32 -> bf16 packed, ds_store_b64 ----
        #pragma unroll 4
        for (int rep = 0; rep < 8; ++rep) {
            const int r = rrA + rep * 16;
            const float4 f =
                *(const float4*)&x[(size_t)(wgM * 128 + r) * D_DIM + k0 + kcA];
            uint2 p;
            p.x = pack2bf(f.x, f.y);
            p.y = pack2bf(f.z, f.w);
            *(uint2*)&ldsA[r * BKP + kcA] = p;
        }
        // ---- stage B tile (64 x 256) -> bf16 transposed [n][k], b32 stores ----
        #pragma unroll 4
        for (int kk = 0; kk < 16; ++kk) {
            const int k = kbB + kk * 4;
            const float2 f0 =
                *(const float2*)&W1[(size_t)(k0 + k) * H_DIM + wgN * 256 + n2];
            const float2 f1 =
                *(const float2*)&W1[(size_t)(k0 + k + 1) * H_DIM + wgN * 256 + n2];
            *(unsigned int*)&ldsB[n2 * BKP + k]       = pack2bf(f0.x, f1.x);
            *(unsigned int*)&ldsB[(n2 + 1) * BKP + k] = pack2bf(f0.y, f1.y);
        }
        // prefetch next K-tile into cache hierarchy (global_prefetch_b8)
        if (k0 + BK < D_DIM) {
            __builtin_prefetch(&x[(size_t)(wgM * 128 + rrA) * D_DIM + k0 + BK + kcA], 0, 1);
            __builtin_prefetch(&W1[(size_t)(k0 + BK + kbB) * H_DIM + wgN * 256 + n2], 0, 1);
        }
        __syncthreads();

        #pragma unroll
        for (int ks = 0; ks < BK; ks += 32) {
            Frag a[4], b[4];
            #pragma unroll
            for (int i = 0; i < 4; ++i) {
                const unsigned short* p =
                    &ldsA[(wr * 64 + i * 16 + lm) * BKP + ks + hi * 8];
                a[i].u[0] = *(const uint4*)p;
                a[i].u[1] = *(const uint4*)(p + 16);
            }
            #pragma unroll
            for (int j = 0; j < 4; ++j) {
                const unsigned short* p =
                    &ldsB[(wc * 64 + j * 16 + lm) * BKP + ks + hi * 8];
                b[j].u[0] = *(const uint4*)p;
                b[j].u[1] = *(const uint4*)(p + 16);
            }
            #pragma unroll
            for (int i = 0; i < 4; ++i)
                #pragma unroll
                for (int j = 0; j < 4; ++j)
                    acc[i][j] = __builtin_amdgcn_wmma_f32_16x16x32_bf16(
                        false, a[i].v, false, b[j].v, (short)0, acc[i][j],
                        false, false);
        }
        __syncthreads();
    }

    // epilogue: fold b1 + BN, relu, bf16 store
    #pragma unroll
    for (int j = 0; j < 4; ++j) {
        const int gn = wgN * 256 + wc * 64 + j * 16 + lm;
        const float sc = gam[gn] * rsqrtf(var[gn] + 1e-5f);
        const float sh = (b1[gn] - mu[gn]) * sc + bet[gn];
        #pragma unroll
        for (int i = 0; i < 4; ++i) {
            const int rbase = wgM * 128 + wr * 64 + i * 16 + hi * 8;
            #pragma unroll
            for (int v = 0; v < 8; ++v) {
                float val = fmaxf(acc[i][j][v] * sc + sh, 0.0f);
                z[(size_t)(rbase + v) * H_DIM + gn] = f2bf(val);
            }
        }
    }
}

// ---------------------------------------------------------------------------
// Kernel 2: in-place running max over s=0..7 (rows b*8+s), 8 bf16 per thread.
// ---------------------------------------------------------------------------
__global__ __launch_bounds__(256) void cummax_k(unsigned short* __restrict__ z)
{
    const size_t gid = (size_t)blockIdx.x * 256 + threadIdx.x; // 2048*512 threads
    const int bidx = (int)(gid >> 9);
    const int hq   = (int)(gid & 511) << 3;
    unsigned short* base = z + (size_t)bidx * 8 * H_DIM + hq;

    union U { uint4 u; unsigned short h[8]; } run, cur;
    run.u = *(const uint4*)base;
    #pragma unroll
    for (int s = 1; s < 8; ++s) {
        cur.u = *(const uint4*)(base + (size_t)s * H_DIM);
        #pragma unroll
        for (int e = 0; e < 8; ++e) {
            const float a = __uint_as_float((unsigned int)run.h[e] << 16);
            const float c = __uint_as_float((unsigned int)cur.h[e] << 16);
            if (c > a) run.h[e] = cur.h[e];
        }
        *(uint4*)(base + (size_t)s * H_DIM) = run.u;
    }
}

// ---------------------------------------------------------------------------
// Kernel 3: logits[b,s,c] = cm[b,s,:] @ Wc[s,:,:] + bc[s,:]; also last_out.
// Per WG: 128 rows (b), all 208 padded cols, K=4096. 8 waves x (16 rows, 13 tiles).
// ---------------------------------------------------------------------------
__global__ __launch_bounds__(256) void gemm2_k(
    const unsigned short* __restrict__ cm, const float* __restrict__ Wc,
    const float* __restrict__ bc, float* __restrict__ out)
{
    __shared__ unsigned short ldsA[128 * BKP];
    __shared__ unsigned short ldsB[208 * BKP];

    const int t    = threadIdx.x;
    const int wgM  = blockIdx.y;      // 0..15
    const int s    = blockIdx.z;      // 0..7
    const int wave = t >> 5, lane = t & 31, hi = lane >> 4, lm = lane & 15;

    v8f acc[13];
    #pragma unroll
    for (int j = 0; j < 13; ++j)
        #pragma unroll
        for (int e = 0; e < 8; ++e) acc[j][e] = 0.0f;

    for (int k0 = 0; k0 < H_DIM; k0 += BK) {
        // stage A (128 x 64 bf16), rows are cm row (b*8+s)
        #pragma unroll 4
        for (int rep = 0; rep < 8; ++rep) {
            const int l  = t + rep * 256;
            const int r  = l >> 4;
            const int kq = (l & 15) * 4;
            const uint2 d = *(const uint2*)&cm[
                ((size_t)(wgM * 128 + r) * 8 + s) * H_DIM + k0 + kq];
            *(uint2*)&ldsA[r * BKP + kq] = d;
        }
        // stage B: Wc[s, k0+k, c] fp32 -> bf16 transposed [c][k], pad c to 208
        if (t < 208) {
            const float* wp = Wc + ((size_t)s * H_DIM + k0) * C_DIM + t;
            #pragma unroll 4
            for (int k = 0; k < BK; ++k) {
                const float v = (t < C_DIM) ? wp[(size_t)k * C_DIM] : 0.0f;
                ldsB[t * BKP + k] = f2bf(v);
            }
        }
        __syncthreads();

        #pragma unroll
        for (int ks = 0; ks < BK; ks += 32) {
            Frag a;
            {
                const unsigned short* p = &ldsA[(wave * 16 + lm) * BKP + ks + hi * 8];
                a.u[0] = *(const uint4*)p;
                a.u[1] = *(const uint4*)(p + 16);
            }
            #pragma unroll
            for (int j = 0; j < 13; ++j) {
                Frag bf;
                const unsigned short* p = &ldsB[(j * 16 + lm) * BKP + ks + hi * 8];
                bf.u[0] = *(const uint4*)p;
                bf.u[1] = *(const uint4*)(p + 16);
                acc[j] = __builtin_amdgcn_wmma_f32_16x16x32_bf16(
                    false, a.v, false, bf.v, (short)0, acc[j], false, false);
            }
        }
        __syncthreads();
    }

    // epilogue
    #pragma unroll
    for (int j = 0; j < 13; ++j) {
        const int gn = j * 16 + lm;
        if (gn < C_DIM) {
            const float bcv = bc[s * C_DIM + gn];
            #pragma unroll
            for (int v = 0; v < 8; ++v) {
                const int bb = wgM * 128 + wave * 16 + hi * 8 + v;
                const float val = acc[j][v] + bcv;
                out[(size_t)(bb * 8 + s) * C_DIM + gn] = val;
                if (s == 7)
                    out[(size_t)M_DIM * C_DIM + (size_t)bb * C_DIM + gn] = val;
            }
        }
    }
}

// ---------------------------------------------------------------------------
extern "C" void kernel_launch(void* const* d_in, const int* in_sizes, int n_in,
                              void* d_out, int out_size, void* d_ws, size_t ws_size,
                              hipStream_t stream)
{
    const float* x    = (const float*)d_in[0];
    const float* W1   = (const float*)d_in[1];
    const float* b1   = (const float*)d_in[2];
    const float* gam  = (const float*)d_in[3];
    const float* bet  = (const float*)d_in[4];
    const float* mu   = (const float*)d_in[5];
    const float* var  = (const float*)d_in[6];
    const float* Wc   = (const float*)d_in[7];
    const float* bc   = (const float*)d_in[8];

    unsigned short* z = (unsigned short*)d_ws;   // 16384*4096 bf16 = 128 MB

    dim3 g1(H_DIM / 256, M_DIM / 128);           // (16, 128)
    gemm1_bn_relu<<<g1, 256, 0, stream>>>(x, W1, b1, gam, bet, mu, var, z);

    cummax_k<<<(B_DIM * (H_DIM / 8)) / 256, 256, 0, stream>>>(z);

    dim3 g2(1, B_DIM / 128, S_DIM);              // (1, 16, 8)
    gemm2_k<<<g2, 256, 0, stream>>>(z, Wc, bc, (float*)d_out);
}
